// CausalMultiHeadSelfAttention_21311627723073
// MI455X (gfx1250) — matrix operand order repaired and checked by reading, not compiled, not run on hardware
//
#include <hip/hip_runtime.h>
#include <math.h>

typedef __attribute__((ext_vector_type(16))) __bf16 v16bf;
typedef __attribute__((ext_vector_type(8)))  float  v8f;

#define BDIM   4
#define SEQ    2048
#define DMODEL 1024
#define NHEAD  16
#define DHEAD  64

__device__ inline v8f wmma_bf16(v16bf a, v16bf b, v8f c) {
  // D = A(16x32 bf16) * B(32x16 bf16) + C(16x16 f32)
  return __builtin_amdgcn_wmma_f32_16x16x32_bf16(false, a, false, b, (short)0, c,
                                                 false, false);
}

// Load a 16-element bf16 fragment as two 16-byte chunks: p and p+step (elements).
__device__ inline v16bf ld_frag2(const __bf16* p, int step) {
  union { v16bf v; uint4 q[2]; } u;
  u.q[0] = *(const uint4*)(p);
  u.q[1] = *(const uint4*)(p + step);
  return u.v;
}

// ---------------------------------------------------------------------------
// Kernel 1: QKV projection. C[m,f] = sum_e X[m,e] * Wqkv[f,e]
// Epilogue scatters into Q[b,h,s,d], K[b,h,s,d] (pre-RoPE) and Vt[b,h,d,s].
// ---------------------------------------------------------------------------
__global__ __launch_bounds__(256) void qkv_gemm(const float* __restrict__ X,
                                                const float* __restrict__ W,
                                                __bf16* __restrict__ Qr,
                                                __bf16* __restrict__ Kr,
                                                __bf16* __restrict__ Vt) {
  __shared__ alignas(16) __bf16 As[128][48];
  __shared__ alignas(16) __bf16 Bs[128][48];
  const int m0 = blockIdx.y * 128;
  const int n0 = blockIdx.x * 128;
  const int tid  = threadIdx.x;
  const int wave = tid >> 5, lane = tid & 31;
  const int wm = wave >> 2, wn = wave & 3;
  const int lr    = lane & 15;
  const int kb8   = (lane < 16) ? 0 : 8;
  const int kb16  = (lane < 16) ? 0 : 16;
  const int rhalf = (lane < 16) ? 0 : 8;

  v8f c[4][2];
  for (int mi = 0; mi < 4; ++mi)
    for (int ni = 0; ni < 2; ++ni)
      for (int e = 0; e < 8; ++e) c[mi][ni][e] = 0.f;

  for (int k0 = 0; k0 < DMODEL; k0 += 32) {
    // Stage 128x32 tiles of A (from X) and B (from Wqkv rows), f32 -> bf16.
    for (int i = 0; i < 4; ++i) {
      int v = tid + i * 256;                 // 0..1023 float4 chunks
      int row = v >> 3, col = (v & 7) << 2;
      float4 fa = *(const float4*)(X + (size_t)(m0 + row) * DMODEL + k0 + col);
      union { __bf16 h[4]; uint2 u; } pa;
      pa.h[0] = (__bf16)fa.x; pa.h[1] = (__bf16)fa.y;
      pa.h[2] = (__bf16)fa.z; pa.h[3] = (__bf16)fa.w;
      *(uint2*)&As[row][col] = pa.u;
      float4 fb = *(const float4*)(W + (size_t)(n0 + row) * DMODEL + k0 + col);
      union { __bf16 h[4]; uint2 u; } pb;
      pb.h[0] = (__bf16)fb.x; pb.h[1] = (__bf16)fb.y;
      pb.h[2] = (__bf16)fb.z; pb.h[3] = (__bf16)fb.w;
      *(uint2*)&Bs[row][col] = pb.u;
    }
    __syncthreads();
    v16bf a[4], bb[2];
    for (int mi = 0; mi < 4; ++mi)
      a[mi] = ld_frag2(&As[wm * 64 + mi * 16 + lr][kb16], 8);
    for (int ni = 0; ni < 2; ++ni)
      bb[ni] = ld_frag2(&Bs[wn * 32 + ni * 16 + lr][kb16], 8);
    for (int mi = 0; mi < 4; ++mi)
      for (int ni = 0; ni < 2; ++ni)
        c[mi][ni] = wmma_bf16(a[mi], bb[ni], c[mi][ni]);
    __syncthreads();
  }

  // Epilogue. A 16-wide fragment group never crosses a q/k/v (1024) or head
  // (64) boundary, so which/h/dd are uniform per ni -> hoist the decode.
  for (int ni = 0; ni < 2; ++ni) {
    const int fstart = n0 + wn * 32 + ni * 16;
    const int which  = fstart >> 10;          // 0=Q 1=K 2=V (uniform)
    const int h      = (fstart >> 6) & (NHEAD - 1);
    const int dd     = (fstart & 63) + lr;    // lane's d-index within the head
    if (which < 2) {
      __bf16* __restrict__ dst = (which == 0) ? Qr : Kr;
      for (int mi = 0; mi < 4; ++mi)
        for (int e = 0; e < 8; ++e) {
          int m  = m0 + wm * 64 + mi * 16 + e + rhalf;
          int bi = m >> 11, sp = m & (SEQ - 1);
          int hs = bi * NHEAD + h;
          dst[((size_t)hs * SEQ + sp) * DHEAD + dd] = (__bf16)c[mi][ni][e];
        }
    } else {
      for (int mi = 0; mi < 4; ++mi)
        for (int e = 0; e < 8; ++e) {
          int m  = m0 + wm * 64 + mi * 16 + e + rhalf;
          int bi = m >> 11, sp = m & (SEQ - 1);
          int hs = bi * NHEAD + h;
          Vt[((size_t)hs * DHEAD + dd) * SEQ + sp] = (__bf16)c[mi][ni][e];
        }
    }
  }
}

// ---------------------------------------------------------------------------
// Kernel 2: RoPE in place on Q and K; fold softmax scale 1/sqrt(64) into Q.
// One 32-thread group per (b,h,s) row; thread t handles the pair (2t, 2t+1).
// ---------------------------------------------------------------------------
__global__ __launch_bounds__(256) void rope_kernel(__bf16* __restrict__ Qr,
                                                   __bf16* __restrict__ Kr,
                                                   const int* __restrict__ pos) {
  int row = blockIdx.x * 8 + (threadIdx.x >> 5);
  int t = threadIdx.x & 31;
  int sp = row & (SEQ - 1);
  float p = (float)pos[sp];
  float freq = __powf(10000.f, -((float)(2 * t) / (float)DHEAD));
  float ang = p * freq;
  float sn = __sinf(ang), cs = __cosf(ang);
  size_t off = (size_t)row * DHEAD + 2 * t;
  float q1 = (float)Qr[off], q2 = (float)Qr[off + 1];
  Qr[off]     = (__bf16)((q1 * cs - q2 * sn) * 0.125f);
  Qr[off + 1] = (__bf16)((q1 * sn + q2 * cs) * 0.125f);
  float k1 = (float)Kr[off], k2 = (float)Kr[off + 1];
  Kr[off]     = (__bf16)(k1 * cs - k2 * sn);
  Kr[off + 1] = (__bf16)(k1 * sn + k2 * cs);
}

// ---------------------------------------------------------------------------
// Kernel 3: causal flash attention. Block = 128 queries of one (b,h); each
// wave owns 16 queries and streams 32-key tiles with online softmax.
// ---------------------------------------------------------------------------
__global__ __launch_bounds__(256) void attn_kernel(const __bf16* __restrict__ Qr,
                                                   const __bf16* __restrict__ Kr,
                                                   const __bf16* __restrict__ Vt,
                                                   __bf16* __restrict__ Yw) {
  __shared__ alignas(16) __bf16 Pst[8][16][48];   // per-wave P staging
  const int bh = blockIdx.y;
  const int q0 = blockIdx.x * 128;
  const int tid  = threadIdx.x;
  const int wave = tid >> 5, lane = tid & 31;
  const int lr    = lane & 15;
  const int kb8   = (lane < 16) ? 0 : 8;
  const int kb16  = (lane < 16) ? 0 : 16;
  const int rhalf = (lane < 16) ? 0 : 8;
  const int qbase = q0 + wave * 16;

  // Q A-fragments (d 0..31, 32..63), held in registers for the whole loop.
  const __bf16* qp = Qr + ((size_t)bh * SEQ + qbase + lr) * DHEAD;
  v16bf aq0 = ld_frag2(qp + kb16, 8);
  v16bf aq1 = ld_frag2(qp + 32 + kb16, 8);

  float mo[8], li[8];
  v8f acc[4];
  for (int e = 0; e < 8; ++e) { mo[e] = -INFINITY; li[e] = 0.f; }
  for (int t4 = 0; t4 < 4; ++t4)
    for (int e = 0; e < 8; ++e) acc[t4][e] = 0.f;

  const int nkt = (qbase + 16 + 31) >> 5;          // causal: keys < qbase+16
  for (int kt = 0; kt < nkt; ++kt) {
    const int kbeg = kt << 5;
    // S = Q * K^T for 32 keys -> two 16x16 score fragments.
    v8f s0, s1;
    {
      const __bf16* kp0 = Kr + ((size_t)bh * SEQ + kbeg + lr) * DHEAD;
      v16bf b00 = ld_frag2(kp0 + kb16, 8);
      v16bf b01 = ld_frag2(kp0 + 32 + kb16, 8);
      v8f z; for (int e = 0; e < 8; ++e) z[e] = 0.f;
      s0 = wmma_bf16(aq0, b00, z);
      s0 = wmma_bf16(aq1, b01, s0);
      const __bf16* kp1 = Kr + ((size_t)bh * SEQ + kbeg + 16 + lr) * DHEAD;
      v16bf b10 = ld_frag2(kp1 + kb16, 8);
      v16bf b11 = ld_frag2(kp1 + 32 + kb16, 8);
      s1 = wmma_bf16(aq0, b10, z);
      s1 = wmma_bf16(aq1, b11, s1);
    }
    const int key0 = kbeg + lr, key1 = key0 + 16;
    float scl[8];
    for (int e = 0; e < 8; ++e) {
      int qr = qbase + e + rhalf;
      float v0 = (key0 <= qr) ? s0[e] : -INFINITY;
      float v1 = (key1 <= qr) ? s1[e] : -INFINITY;
      float r = fmaxf(v0, v1);
      for (int off = 1; off < 16; off <<= 1) r = fmaxf(r, __shfl_xor(r, off, 32));
      float mn = fmaxf(mo[e], r);
      float p0 = __expf(v0 - mn);
      float p1 = __expf(v1 - mn);
      float rs = p0 + p1;
      for (int off = 1; off < 16; off <<= 1) rs += __shfl_xor(rs, off, 32);
      float sc = __expf(mo[e] - mn);
      li[e] = li[e] * sc + rs;
      mo[e] = mn;
      scl[e] = sc;
      Pst[wave][e + rhalf][lr]      = (__bf16)p0;   // C-frag -> A-frag via LDS
      Pst[wave][e + rhalf][16 + lr] = (__bf16)p1;
    }
    for (int t4 = 0; t4 < 4; ++t4)
      for (int e = 0; e < 8; ++e) acc[t4][e] *= scl[e];

    v16bf pf = ld_frag2(&Pst[wave][lr][kb16], 8);   // P as A-fragment
    for (int t4 = 0; t4 < 4; ++t4) {
      const __bf16* vp =
          Vt + ((size_t)bh * DHEAD + t4 * 16 + lr) * SEQ + kbeg + kb16;
      v16bf vf = ld_frag2(vp, 8);                   // V^T rows are contiguous
      acc[t4] = wmma_bf16(pf, vf, acc[t4]);
    }
  }

  // Normalize (one reciprocal per row) and write Y in [b,s,d_model] bf16.
  float inv[8];
  for (int e = 0; e < 8; ++e) inv[e] = __frcp_rn(li[e]);
  const int bi = bh >> 4, h = bh & 15;
  __bf16* __restrict__ yp =
      Yw + (size_t)(bi * SEQ + qbase + rhalf) * DMODEL + h * DHEAD + lr;
  for (int t4 = 0; t4 < 4; ++t4)
    for (int e = 0; e < 8; ++e)
      yp[(size_t)e * DMODEL + t4 * 16] = (__bf16)(acc[t4][e] * inv[e]);
}

// ---------------------------------------------------------------------------
// Kernel 4: output projection. Out[m,f] = sum_e Y[m,e] * Wo[f,e], f32 out.
// ---------------------------------------------------------------------------
__global__ __launch_bounds__(256) void out_gemm(const __bf16* __restrict__ Y,
                                                const float* __restrict__ W,
                                                float* __restrict__ Out) {
  __shared__ alignas(16) __bf16 As[128][48];
  __shared__ alignas(16) __bf16 Bs[128][48];
  const int m0 = blockIdx.y * 128;
  const int n0 = blockIdx.x * 128;
  const int tid  = threadIdx.x;
  const int wave = tid >> 5, lane = tid & 31;
  const int wm = wave >> 2, wn = wave & 3;
  const int lr    = lane & 15;
  const int kb8   = (lane < 16) ? 0 : 8;
  const int kb16  = (lane < 16) ? 0 : 16;
  const int rhalf = (lane < 16) ? 0 : 8;

  v8f c[4][2];
  for (int mi = 0; mi < 4; ++mi)
    for (int ni = 0; ni < 2; ++ni)
      for (int e = 0; e < 8; ++e) c[mi][ni][e] = 0.f;

  for (int k0 = 0; k0 < DMODEL; k0 += 32) {
    for (int i = 0; i < 2; ++i) {        // A already bf16: 512 x (8 bf16)
      int v = tid + i * 256;
      int row = v >> 2, col = (v & 3) << 3;
      uint4 d = *(const uint4*)(Y + (size_t)(m0 + row) * DMODEL + k0 + col);
      *(uint4*)&As[row][col] = d;
    }
    for (int i = 0; i < 4; ++i) {        // B f32 -> bf16
      int v = tid + i * 256;
      int row = v >> 3, col = (v & 7) << 2;
      float4 fb = *(const float4*)(W + (size_t)(n0 + row) * DMODEL + k0 + col);
      union { __bf16 h[4]; uint2 u; } pb;
      pb.h[0] = (__bf16)fb.x; pb.h[1] = (__bf16)fb.y;
      pb.h[2] = (__bf16)fb.z; pb.h[3] = (__bf16)fb.w;
      *(uint2*)&Bs[row][col] = pb.u;
    }
    __syncthreads();
    v16bf a[4], bb[2];
    for (int mi = 0; mi < 4; ++mi)
      a[mi] = ld_frag2(&As[wm * 64 + mi * 16 + lr][kb16], 8);
    for (int ni = 0; ni < 2; ++ni)
      bb[ni] = ld_frag2(&Bs[wn * 32 + ni * 16 + lr][kb16], 8);
    for (int mi = 0; mi < 4; ++mi)
      for (int ni = 0; ni < 2; ++ni)
        c[mi][ni] = wmma_bf16(a[mi], bb[ni], c[mi][ni]);
    __syncthreads();
  }
  // Contiguous f32 stores: 16 lanes cover 64 consecutive bytes per row.
  for (int mi = 0; mi < 4; ++mi)
    for (int e = 0; e < 8; ++e) {
      int m = m0 + wm * 64 + mi * 16 + e + rhalf;
      float* __restrict__ op = Out + (size_t)m * DMODEL + n0 + wn * 32 + lr;
      op[0]  = c[mi][0][e];
      op[16] = c[mi][1][e];
    }
}

extern "C" void kernel_launch(void* const* d_in, const int* in_sizes, int n_in,
                              void* d_out, int out_size, void* d_ws, size_t ws_size,
                              hipStream_t stream) {
  const float* x    = (const float*)d_in[0];
  const float* wqkv = (const float*)d_in[1];
  const float* wo   = (const float*)d_in[2];
  const int*   pos  = (const int*)d_in[3];
  float* out = (float*)d_out;

  const size_t seg = (size_t)BDIM * NHEAD * SEQ * DHEAD;  // 8388608 elements
  __bf16* Qr = (__bf16*)d_ws;
  __bf16* Kr = Qr + seg;
  __bf16* Vt = Kr + seg;
  __bf16* Yw = Vt + seg;   // 4 * 16 MB = 64 MB of workspace

  qkv_gemm<<<dim3(24, 64), 256, 0, stream>>>(x, wqkv, Qr, Kr, Vt);
  rope_kernel<<<dim3((BDIM * NHEAD * SEQ) / 8), 256, 0, stream>>>(Qr, Kr, pos);
  attn_kernel<<<dim3(SEQ / 128, BDIM * NHEAD), 256, 0, stream>>>(Qr, Kr, Vt, Yw);
  out_gemm<<<dim3(8, 64), 256, 0, stream>>>(Yw, wo, out);
}
